// CorrFeature_12120397710172
// MI455X (gfx1250) — matrix operand order repaired and checked by reading, not compiled, not run on hardware
//
#include <hip/hip_runtime.h>
#include <hip/hip_bf16.h>

// ---------------------------------------------------------------------------
// FlowNet-C front end for MI455X (gfx1250, wave32):
//   siamese 3-layer conv encoder (conv+BN+leakyReLU) via implicit-GEMM with
//   v_wmma_f32_16x16x32_f16. Templated layer geometry so the wave/tile
//   partition exactly covers Cout (no guards around WMMA). Double-buffered
//   LDS staging (16B-aligned, bank-padded rows -> ds_load/store_b128),
//   packed im2col decode table, branch-free boundary path, unconditional
//   interior fast path. Correlation (441 displacements) runs on VALU:
//   ~3% of FLOPs, L2-resident, not expressible as a shared-operand GEMM.
// ---------------------------------------------------------------------------

typedef __attribute__((ext_vector_type(16))) _Float16 v16h;
typedef __attribute__((ext_vector_type(8)))  float    v8f;

#define LEAK   0.1f
#define BN_EPS 1e-5f

union Frag16 { v16h v; unsigned u[8]; };

// ---- convert f32 input (N,6,H,W) into two f16 tensors (N,3,H,W) ------------
__global__ void split_convert_kernel(const float* __restrict__ x,
                                     _Float16* __restrict__ x1h,
                                     _Float16* __restrict__ x2h) {
  const int HW  = 448 * 448;
  const int CHW = 3 * HW;
  int i = blockIdx.x * blockDim.x + threadIdx.x;
  if (i >= 8 * CHW) return;
  int n = i / CHW;
  int r = i - n * CHW;
  x1h[i] = (_Float16)x[(size_t)n * 6 * HW + r];
  x2h[i] = (_Float16)x[(size_t)n * 6 * HW + CHW + r];
}

// ---- pack OIHW f32 weights into [co][Kpad] f16 (im2col K order, 0-padded) --
__global__ void pack_weights_kernel(const float* __restrict__ w,
                                    _Float16* __restrict__ wp,
                                    int Cout, int K, int Kpad) {
  int i = blockIdx.x * blockDim.x + threadIdx.x;
  if (i >= Cout * Kpad) return;
  int co = i / Kpad;
  int k  = i - co * Kpad;
  wp[i] = (k < K) ? (_Float16)w[co * K + k] : (_Float16)0.0f;
}

// ---- implicit-GEMM conv + BN + leaky ReLU ----------------------------------
template <int WAVES, int TILES,
          int CIN, int HIN, int WIN,
          int KSZ, int STRIDE, int PAD,
          int K, int KPAD, int COUT, int HOUT, int WOUT>
__global__ __launch_bounds__(WAVES * 32)
void conv_bn_lrelu_wmma(const _Float16* __restrict__ in,
                        const _Float16* __restrict__ wp,
                        const float* __restrict__ gam,
                        const float* __restrict__ bet,
                        const float* __restrict__ mu,
                        const float* __restrict__ var,
                        _Float16* __restrict__ out) {
  static_assert(WAVES * TILES * 16 == COUT, "tiles must exactly cover Cout");
  static_assert(KPAD % 32 == 0, "KPAD must be a multiple of 32");

  constexpr int NT      = WAVES * 32;   // threads per block
  constexpr int GRP     = WAVES / 2;    // k-groups per pixel when staging
  constexpr int KK_PER  = 64 / WAVES;   // k-elements staged per thread
  constexpr int HINWIN  = HIN * WIN;
  constexpr int KSQ     = KSZ * KSZ;
  constexpr int NCHUNK  = KPAD / 32;
  constexpr int BROW    = 40;           // LDS row stride (f16): 80B, 16B
                                        // aligned + conflict-free banking

  __shared__ __align__(16) _Float16 sB[2][64 * BROW]; // 2 x 5 KB
  __shared__ int sKd[KPAD];             // packed im2col decode per kidx

  const int tid  = threadIdx.x;
  const int lane = tid & 31;
  const int wave = tid >> 5;
  const int n    = blockIdx.y;
  const int pix0 = blockIdx.x * 64;

  // ---- packed decode table: [31]=invalid, [30:27]=kh, [26:23]=kw,
  //      [22:0] = ci*HIN*WIN + kh*WIN + kw ----
  for (int k0 = tid; k0 < KPAD; k0 += NT) {
    int word = -1;
    if (k0 < K) {
      int ci = k0 / KSQ;
      int rr = k0 - ci * KSQ;
      int kh = rr / KSZ;
      int kw = rr - kh * KSZ;
      word = (kh << 27) | (kw << 23) | (ci * HINWIN + kh * WIN + kw);
    }
    sKd[k0] = word;
  }

  // staging coordinates: this thread gathers KK_PER k-values for one pixel
  const int p_l  = tid / GRP;                       // 0..63
  const int kgrp = tid % GRP;
  const int gpix_s = pix0 + p_l;
  const int oh  = gpix_s / WOUT;
  const int ow  = gpix_s - oh * WOUT;
  const int ih0 = oh * STRIDE - PAD;
  const int iw0 = ow * STRIDE - PAD;
  const int base0 = ih0 * WIN + iw0;                // may be negative
  const _Float16* inN = in + (size_t)n * CIN * HINWIN;
  // receptive field fully inside the input? (uniform per thread)
  const bool interior = (ih0 >= 0) && (ih0 + KSZ <= HIN) &&
                        (iw0 >= 0) && (iw0 + KSZ <= WIN);

  auto stageChunk = [&](int kc, _Float16* __restrict__ dst) {
    // vectorized decode-word fetch: KK_PER contiguous, 16B-aligned ints
    int wbuf[KK_PER];
    #pragma unroll
    for (int q = 0; q < KK_PER / 4; ++q) {
      int4 wv = *(const int4*)&sKd[kc * 32 + kgrp * KK_PER + q * 4];
      wbuf[q * 4 + 0] = wv.x;
      wbuf[q * 4 + 1] = wv.y;
      wbuf[q * 4 + 2] = wv.z;
      wbuf[q * 4 + 3] = wv.w;
    }
    union { _Float16 h[KK_PER]; uint4 q[KK_PER / 8]; } sb;
    const bool fullChunk = (kc * 32 + 32 <= K);     // uniform per chunk
    if (interior && fullChunk) {
      // fast path: unconditional gather
      #pragma unroll
      for (int j = 0; j < KK_PER; ++j)
        sb.h[j] = inN[base0 + (wbuf[j] & 0x7FFFFF)];
    } else {
      // branch-free checked gather (cndmask, no exec-mask nesting)
      #pragma unroll
      for (int j = 0; j < KK_PER; ++j) {
        int word = wbuf[j];
        int kh = (word >> 27) & 15;
        int kw = (word >> 23) & 15;
        bool ok = (word >= 0) &
                  ((unsigned)(ih0 + kh) < (unsigned)HIN) &
                  ((unsigned)(iw0 + kw) < (unsigned)WIN);
        int idx = ok ? (base0 + (word & 0x7FFFFF)) : 0;
        _Float16 val = inN[idx];
        sb.h[j] = ok ? val : (_Float16)0.0f;
      }
    }
    // 16B-aligned vector stores into LDS (ds_store_b128)
    uint4* drow = (uint4*)(dst + p_l * BROW + kgrp * KK_PER);
    #pragma unroll
    for (int q = 0; q < KK_PER / 8; ++q) drow[q] = sb.q[q];
  };

  v8f acc[TILES][4];
  #pragma unroll
  for (int t = 0; t < TILES; ++t)
    #pragma unroll
    for (int s = 0; s < 4; ++s)
      #pragma unroll
      for (int r = 0; r < 8; ++r) acc[t][s][r] = 0.0f;

  const int mrow = lane & 15;        // A: M index
  const int half = lane >> 4;        // A/B: K-half select
  const int npix = lane & 15;        // B: N index
  const int koff = half * 16;        // unused; B's half offset is half * 8

  __syncthreads();                   // decode table ready
  stageChunk(0, sB[0]);
  __syncthreads();

  for (int kc = 0; kc < NCHUNK; ++kc) {
    const int cur = kc & 1;
    if (kc + 1 < NCHUNK) stageChunk(kc + 1, sB[cur ^ 1]);

    // A fragments: CDNA5 16-bit A layout (K split 0-7/16-23 vs 8-15/24-31)
    Frag16 a[TILES];
    #pragma unroll
    for (int t = 0; t < TILES; ++t) {
      const int coTile = wave + WAVES * t;          // always < COUT/16
      const _Float16* wrow =
          wp + (size_t)(coTile * 16 + mrow) * KPAD + kc * 32 + half * 8;
      #pragma unroll
      for (int vv = 0; vv < 8; ++vv) {
        int kb = (vv < 4) ? (2 * vv) : (16 + 2 * (vv - 4));
        a[t].u[vv] = *(const unsigned*)(wrow + kb);
      }
    }
    // preload all 4 B fragments, then issue 4*TILES back-to-back WMMAs
    Frag16 b[4];
    #pragma unroll
    for (int s = 0; s < 4; ++s) {
      const _Float16* brow = &sB[cur][(s * 16 + npix) * BROW + half * 8];
      #pragma unroll
      for (int vv = 0; vv < 8; ++vv)
        b[s].u[vv] = *(const unsigned*)(brow + ((vv < 4) ? (2 * vv) : (16 + 2 * (vv - 4))));
    }
    #pragma unroll
    for (int s = 0; s < 4; ++s)
      #pragma unroll
      for (int t = 0; t < TILES; ++t)
        acc[t][s] = __builtin_amdgcn_wmma_f32_16x16x32_f16(
            false, a[t].v, false, b[s].v, (short)0, acc[t][s], false, false);

    __syncthreads();
  }

  // fused BN (scale/shift) + leaky ReLU epilogue
  constexpr size_t HWo = (size_t)HOUT * WOUT;
  #pragma unroll
  for (int t = 0; t < TILES; ++t) {
    const int coTile = wave + WAVES * t;
    float scale[8], shift[8];
    #pragma unroll
    for (int r = 0; r < 8; ++r) {
      int co = coTile * 16 + r + 8 * half;
      float inv = gam[co] * rsqrtf(var[co] + BN_EPS);
      scale[r] = inv;
      shift[r] = bet[co] - mu[co] * inv;
    }
    #pragma unroll
    for (int s = 0; s < 4; ++s) {
      int gp = pix0 + s * 16 + npix;
      #pragma unroll
      for (int r = 0; r < 8; ++r) {
        int co = coTile * 16 + r + 8 * half;
        float y = acc[t][s][r] * scale[r] + shift[r];
        y = (y > 0.0f) ? y : LEAK * y;
        out[(size_t)(n * COUT + co) * HWo + gp] = (_Float16)y;
      }
    }
  }
}

// ---- 441-displacement correlation + mean + leaky ReLU ----------------------
// Channel-major layout: adjacent lanes read adjacent pixels -> coalesced.
// Data is L2-resident (fa/fb: 12.8 MB each); 2.8 GMAC total -> VALU is fine.
__global__ __launch_bounds__(256)
void corr_lrelu_kernel(const _Float16* __restrict__ fa,
                       const _Float16* __restrict__ fb,
                       float* __restrict__ out) {
  const int HW = 56 * 56;
  const int ND = 441;
  long long idx = (long long)blockIdx.x * blockDim.x + threadIdx.x;
  const long long total = 8LL * ND * HW;
  if (idx >= total) return;
  int w = (int)(idx % 56);
  int h = (int)((idx / 56) % 56);
  int d = (int)((idx / HW) % ND);
  int n = (int)(idx / ((long long)HW * ND));
  int dy = (d / 21) * 2 - 20;
  int dx = (d % 21) * 2 - 20;
  int h2 = h + dy, w2 = w + dx;
  float acc = 0.0f;
  if (h2 >= 0 && h2 < 56 && w2 >= 0 && w2 < 56) {
    const _Float16* pa = fa + (size_t)n * 256 * HW + h * 56 + w;
    const _Float16* pb = fb + (size_t)n * 256 * HW + h2 * 56 + w2;
    #pragma unroll 8
    for (int c = 0; c < 256; ++c)
      acc += (float)pa[(size_t)c * HW] * (float)pb[(size_t)c * HW];
  }
  acc *= (1.0f / 256.0f);
  out[idx] = (acc > 0.0f) ? acc : LEAK * acc;
}

// ---------------------------------------------------------------------------
extern "C" void kernel_launch(void* const* d_in, const int* in_sizes, int n_in,
                              void* d_out, int out_size, void* d_ws, size_t ws_size,
                              hipStream_t stream) {
  (void)in_sizes; (void)n_in; (void)out_size; (void)ws_size;

  const float* x  = (const float*)d_in[0];
  const float* w1 = (const float*)d_in[1];
  const float* g1 = (const float*)d_in[2];
  const float* b1 = (const float*)d_in[3];
  const float* m1 = (const float*)d_in[4];
  const float* v1 = (const float*)d_in[5];
  const float* w2 = (const float*)d_in[6];
  const float* g2 = (const float*)d_in[7];
  const float* b2 = (const float*)d_in[8];
  const float* m2 = (const float*)d_in[9];
  const float* v2 = (const float*)d_in[10];
  const float* w3 = (const float*)d_in[11];
  const float* g3 = (const float*)d_in[12];
  const float* b3 = (const float*)d_in[13];
  const float* m3 = (const float*)d_in[14];
  const float* v3 = (const float*)d_in[15];

  // workspace carve-out (~125 MB, a1/a2 reused across siamese sides)
  char* ws = (char*)d_ws;
  size_t off = 0;
  auto carve = [&](size_t bytes) -> char* {
    char* p = ws + off;
    off += (bytes + 255) & ~(size_t)255;
    return p;
  };
  _Float16* x1h = (_Float16*)carve((size_t)8 * 3 * 448 * 448 * 2);
  _Float16* x2h = (_Float16*)carve((size_t)8 * 3 * 448 * 448 * 2);
  _Float16* w1p = (_Float16*)carve((size_t)64  * 160  * 2);
  _Float16* w2p = (_Float16*)carve((size_t)128 * 1600 * 2);
  _Float16* w3p = (_Float16*)carve((size_t)256 * 3200 * 2);
  _Float16* a1  = (_Float16*)carve((size_t)8 * 64  * 224 * 224 * 2);
  _Float16* a2  = (_Float16*)carve((size_t)8 * 128 * 112 * 112 * 2);
  _Float16* a3A = (_Float16*)carve((size_t)8 * 256 * 56 * 56 * 2);
  _Float16* a3B = (_Float16*)carve((size_t)8 * 256 * 56 * 56 * 2);

  {
    int nElems = 8 * 3 * 448 * 448;
    split_convert_kernel<<<(nElems + 255) / 256, 256, 0, stream>>>(x, x1h, x2h);
  }
  pack_weights_kernel<<<(64 * 160   + 255) / 256, 256, 0, stream>>>(w1, w1p, 64,  147,  160);
  pack_weights_kernel<<<(128 * 1600 + 255) / 256, 256, 0, stream>>>(w2, w2p, 128, 1600, 1600);
  pack_weights_kernel<<<(256 * 3200 + 255) / 256, 256, 0, stream>>>(w3, w3p, 256, 3200, 3200);

  for (int side = 0; side < 2; ++side) {
    const _Float16* in0 = side ? x2h : x1h;
    _Float16* a3 = side ? a3B : a3A;
    // conv1: 64 couts = 4 waves x 1 tile (128-thread blocks, no idle waves)
    conv_bn_lrelu_wmma<4, 1, 3, 448, 448, 7, 2, 3, 147, 160, 64, 224, 224>
        <<<dim3(224 * 224 / 64, 8, 1), 128, 0, stream>>>(
        in0, w1p, g1, b1, m1, v1, a1);
    // conv2: 128 couts = 8 waves x 1 tile
    conv_bn_lrelu_wmma<8, 1, 64, 224, 224, 5, 2, 2, 1600, 1600, 128, 112, 112>
        <<<dim3(112 * 112 / 64, 8, 1), 256, 0, stream>>>(
        a1, w2p, g2, b2, m2, v2, a2);
    // conv3: 256 couts = 8 waves x 2 tiles (B fragment reused by 2 WMMAs)
    conv_bn_lrelu_wmma<8, 2, 128, 112, 112, 5, 2, 2, 3200, 3200, 256, 56, 56>
        <<<dim3(56 * 56 / 64, 8, 1), 256, 0, stream>>>(
        a2, w3p, g3, b3, m3, v3, a3);
  }

  {
    long long total = 8LL * 441 * 56 * 56;
    int blocks = (int)((total + 255) / 256);
    corr_lrelu_kernel<<<blocks, 256, 0, stream>>>(a3A, a3B, (float*)d_out);
  }
}
